// char_CNN_word_CNN_12756052869584
// MI455X (gfx1250) — matrix operand order repaired and checked by reading, not compiled, not run on hardware
//
#include <hip/hip_runtime.h>
#include <hip/hip_bf16.h>
#include <stdint.h>

// ---------------------------------------------------------------------------
// char-CNN + 4-layer conv1d tagger, WMMA (f16 in / f32 acc) implementation
// for gfx1250 (MI455X, wave32).
//
// Shapes: B=128, S=512, W=16, CE=50, WE=100, CC=200, TAGS=50
// Channel padding: concat embed 150 -> 160 ; conv channels 200 -> 224.
// Activations in f16 [B][S][Cpad] row-major (channel contiguous): matches the
// WMMA B-fragment pair layout for u32/b128 loads and keeps the inter-layer
// working set (~29MB/layer) L2-resident (192MB L2).
//
// Round-2 changes: hoist tile-invariant WMMA operands into register arrays
// (weights per-k in conv kernel; activations per seq tile in out-proj) to cut
// 4x redundant LDS/global fragment traffic, and batch fragment loads ahead of
// the WMMA chains so dep-counter waits amortize instead of one wait per wmma.
// ---------------------------------------------------------------------------

typedef __attribute__((ext_vector_type(16))) _Float16 v16h;
typedef __attribute__((ext_vector_type(8)))  float    v8f;

union U32H2 { uint32_t u; _Float16 h[2]; };

__device__ inline void set_pair(v16h& v, int j, uint32_t u) {
  U32H2 c; c.u = u;
  v[2 * j]     = c.h[0];
  v[2 * j + 1] = c.h[1];
}

// A-fragment (16x32 f16) K base for vgpr j, half 'hi' (lane>=16):
//   j<4: K=2j ; j>=4: K=2j+8 ; upper half adds +8.
__device__ inline int a_kbase(int j, int hi) {
  return 2 * j + ((j >= 4) ? 8 : 0) + (hi ? 8 : 0);
}

__device__ inline v8f wmma_f16(const v16h& a, const v16h& b, const v8f& c) {
  return __builtin_amdgcn_wmma_f32_16x16x32_f16(false, a, false, b,
                                                (short)0, c, false, false);
}

// ---------------------------------------------------------------------------
// Kernel 1: per-word char CNN (conv K=3 SAME over W=16, CE=50->50, relu,
// max over positions) + word-embedding gather + concat into e[B][S][160] f16.
//   channels 0..99 = word emb, 100..149 = char feature, 150..159 = 0
// One wave per word; 8 words per block. WMMA tile: M=16 positions,
// N=16 out-channels (4 tiles), K = 3*50 -> 160 padded (5 chunks of 32).
// ---------------------------------------------------------------------------
__global__ __launch_bounds__(256) void char_word_kernel(
    const int* __restrict__ char_ids,   // [B*S*16]
    const int* __restrict__ text,       // [B*S]
    const float* __restrict__ char_emb, // [100][50]
    const float* __restrict__ cnn_w,    // [50][50][3]
    const float* __restrict__ cnn_b,    // [50]
    const float* __restrict__ word_emb, // [50000][100]
    _Float16* __restrict__ e)           // [B*S][160]
{
  __shared__ _Float16 wlds[64][160];     // B-operand: [co][K], K = k*50+cin
  __shared__ _Float16 chars[8][18][64];  // per-wave: [pos+1][cin], pad rows 0

  const int tid    = threadIdx.x;
  const int wave   = tid >> 5;
  const int lane   = tid & 31;
  const int lane16 = lane & 15;
  const int hi     = lane >> 4;
  const int word   = blockIdx.x * 8 + wave;   // 0..65535

  // --- cooperative weight load (shared by the whole block) ---
  for (int idx = tid; idx < 64 * 160; idx += 256) {
    const int co = idx / 160;
    const int K  = idx % 160;
    _Float16 v = (_Float16)0.f;
    if (co < 50 && K < 150) {
      const int k = K / 50, cin = K % 50;
      v = (_Float16)cnn_w[(co * 50 + cin) * 3 + k];
    }
    wlds[co][K] = v;
  }

  // --- per-wave char-embedding tile (18 rows: pos -1..16, zero-padded) ---
  {
    const int* ids = char_ids + word * 16;
    for (int idx = lane; idx < 18 * 64; idx += 32) {
      const int row = idx >> 6;      // 0..17
      const int cin = idx & 63;      // 0..63
      const int pos = row - 1;
      _Float16 v = (_Float16)0.f;
      if (pos >= 0 && pos < 16 && cin < 50)
        v = (_Float16)char_emb[ids[pos] * 50 + cin];
      chars[wave][row][cin] = v;
    }
  }
  __syncthreads();

  // --- hoist A fragments (im2col of chars) for all 5 K-chunks ---
  v16h afr[5];
#pragma unroll
  for (int c = 0; c < 5; ++c) {
#pragma unroll
    for (int j = 0; j < 8; ++j) {
      const int K = c * 32 + a_kbase(j, hi);
      uint32_t u = 0u;
      if (K < 150) {
        const int k = K / 50, cin = K % 50;  // pair never crosses k boundary
        u = *(const uint32_t*)&chars[wave][lane16 + k][cin];
      }
      set_pair(afr[c], j, u);
    }
  }

  // --- WMMA: out(pos, co) = sum_K  A(pos,K) * Wlds(K,co) ---
  for (int ntile = 0; ntile < 4; ++ntile) {
    // batch all B fragments for this co tile, then run the wmma chain
    v16h bfr[5];
#pragma unroll
    for (int c = 0; c < 5; ++c) {
#pragma unroll
      for (int j = 0; j < 8; ++j) {
        const int K = c * 32 + a_kbase(j, hi);
        set_pair(bfr[c], j, *(const uint32_t*)&wlds[ntile * 16 + lane16][K]);
      }
    }
    v8f acc;
#pragma unroll
    for (int r = 0; r < 8; ++r) acc[r] = 0.f;
#pragma unroll
    for (int c = 0; c < 5; ++c) acc = wmma_f16(afr[c], bfr[c], acc);

    // max over positions: 8 rows in-lane, then lane L <-> L^16
    float m = acc[0];
#pragma unroll
    for (int r = 1; r < 8; ++r) m = fmaxf(m, acc[r]);
    m = fmaxf(m, __shfl_xor(m, 16, 32));

    const int co = ntile * 16 + lane16;
    if (hi == 0 && co < 50) {
      const float bias = cnn_b[co];
      const float r = fmaxf(m + bias, 0.f);        // relu(max + b) == max(relu)
      e[(size_t)word * 160 + 100 + co] = (_Float16)r;
    }
  }

  // --- word embedding + zero padding ---
  {
    const int row = text[word];
    for (int ch = lane; ch < 100; ch += 32)
      e[(size_t)word * 160 + ch] = (_Float16)word_emb[row * 100 + ch];
    if (lane < 10)
      e[(size_t)word * 160 + 150 + lane] = (_Float16)0.f;
  }
}

// ---------------------------------------------------------------------------
// Kernel 2: generic conv1d(K=3, SAME) + bias + relu via WMMA.
//   x: f16 [B][512][CINPAD]   w: f32 [CO][Cin][3]   y: f16 [B][512][224]
// Block = (batch, 16-co tile); 8 waves, each owns 4 seq tiles of 16.
// Loop nest: k outer -> hoist the NCH weight A-fragments into registers once
// per k and reuse across all 4 seq tiles (4x fewer LDS fragment reads); the
// 4 accumulators stay live across the whole reduction.
// Pad co rows get zero weights + zero bias -> relu -> y pad channels == 0.
// ---------------------------------------------------------------------------
template <int CINPAD>
__global__ __launch_bounds__(256) void conv_relu_kernel(
    const _Float16* __restrict__ x,
    const float* __restrict__ w,
    const float* __restrict__ bias,
    _Float16* __restrict__ y,
    int Cin, int CO)
{
  constexpr int NCH = CINPAD / 32;       // 5 (conv1) or 7 (conv2-4)
  __shared__ _Float16 wlds[3][16][CINPAD];  // [k][co_local][cin]

  const int tid    = threadIdx.x;
  const int wave   = tid >> 5;
  const int lane   = tid & 31;
  const int lane16 = lane & 15;
  const int hi     = lane >> 4;

  const int b      = blockIdx.x / 14;
  const int cobase = (blockIdx.x % 14) * 16;

  // cooperative weight panel load (f32 -> f16, zero padded)
  for (int idx = tid; idx < 3 * 16 * CINPAD; idx += 256) {
    const int k   = idx / (16 * CINPAD);
    const int rem = idx % (16 * CINPAD);
    const int col = rem / CINPAD;
    const int cin = rem % CINPAD;
    const int co  = cobase + col;
    _Float16 v = (_Float16)0.f;
    if (co < CO && cin < Cin)
      v = (_Float16)w[(co * Cin + cin) * 3 + k];
    wlds[k][col][cin] = v;
  }
  __syncthreads();

  // bias per accumulator row (co = cobase + r + 8*hi)
  float bv[8];
#pragma unroll
  for (int r = 0; r < 8; ++r) {
    const int co = cobase + r + hi * 8;
    bv[r] = (co < CO) ? bias[co] : 0.f;
  }

  // 4 seq tiles per wave: t0(i) = (wave + 8*i) * 16
  v8f acc[4];
#pragma unroll
  for (int i = 0; i < 4; ++i)
#pragma unroll
    for (int r = 0; r < 8; ++r) acc[i][r] = bv[r];

#pragma unroll
  for (int k = 0; k < 3; ++k) {
    // hoist weight A-fragments for this k (reused by all 4 seq tiles)
    v16h afr[NCH];
#pragma unroll
    for (int c = 0; c < NCH; ++c)
#pragma unroll
      for (int j = 0; j < 8; ++j)
        set_pair(afr[c], j,
                 *(const uint32_t*)&wlds[k][lane16][c * 32 + a_kbase(j, hi)]);

#pragma unroll
    for (int i = 0; i < 4; ++i) {
      const int t0 = (wave + 8 * i) * 16;
      const int t  = t0 + lane16 + k - 1;
      const bool tin = (t >= 0) && (t < 512);
      const _Float16* xrow =
          x + ((size_t)b * 512 + (tin ? t : 0)) * (size_t)CINPAD;

      // batch all B fragments for this (k, tile), then the wmma chain
      v16h bfr[NCH];
#pragma unroll
      for (int c = 0; c < NCH; ++c)
#pragma unroll
        for (int j = 0; j < 8; ++j) {
          const int cin = c * 32 + a_kbase(j, hi);
          set_pair(bfr[c], j, tin ? *(const uint32_t*)&xrow[cin] : 0u);
        }
#pragma unroll
      for (int c = 0; c < NCH; ++c)
        acc[i] = wmma_f16(afr[c], bfr[c], acc[i]);
    }
  }

  // relu + pack pairs -> f16 stores  (co even-aligned)
#pragma unroll
  for (int i = 0; i < 4; ++i) {
    const int t0 = (wave + 8 * i) * 16;
    _Float16* yrow =
        y + ((size_t)b * 512 + t0 + lane16) * 224 + cobase + hi * 8;
#pragma unroll
    for (int p = 0; p < 4; ++p) {
      U32H2 c;
      c.h[0] = (_Float16)fmaxf(acc[i][2 * p], 0.f);
      c.h[1] = (_Float16)fmaxf(acc[i][2 * p + 1], 0.f);
      *(uint32_t*)&yrow[2 * p] = c.u;
    }
  }
}

// ---------------------------------------------------------------------------
// Kernel 3: output projection  out[B][S][50] = h[B][S][200] @ W[200][50] + b
// WMMA: M=16 seq, N=16 tags (4 tiles, 50 valid), K=224 (7 chunks).
// The 7 activation A-fragments are hoisted per seq tile and reused across all
// 4 tag tiles; W^T panel lives in LDS as [tag][cin] (K-pairs contiguous).
// ---------------------------------------------------------------------------
__global__ __launch_bounds__(256) void out_proj_kernel(
    const _Float16* __restrict__ h,
    const float* __restrict__ out_w,  // [200][50]
    const float* __restrict__ out_b,  // [50]
    float* __restrict__ out)          // [B*S*50]
{
  __shared__ _Float16 wT[64][224];

  const int tid    = threadIdx.x;
  const int wave   = tid >> 5;
  const int lane   = tid & 31;
  const int lane16 = lane & 15;
  const int hi     = lane >> 4;
  const int b      = blockIdx.x;

  for (int idx = tid; idx < 64 * 224; idx += 256) {
    const int tag = idx / 224;
    const int cin = idx % 224;
    _Float16 v = (_Float16)0.f;
    if (tag < 50 && cin < 200) v = (_Float16)out_w[cin * 50 + tag];
    wT[tag][cin] = v;
  }
  __syncthreads();

  for (int tt = wave; tt < 32; tt += 8) {
    const int t0 = tt * 16;
    const _Float16* hrow = h + ((size_t)b * 512 + t0 + lane16) * 224;

    // hoist activation A-fragments (reused by all 4 tag tiles)
    v16h afr[7];
#pragma unroll
    for (int c = 0; c < 7; ++c)
#pragma unroll
      for (int j = 0; j < 8; ++j)
        set_pair(afr[c], j,
                 *(const uint32_t*)&hrow[c * 32 + a_kbase(j, hi)]);

    for (int ntile = 0; ntile < 4; ++ntile) {
      const int tag = ntile * 16 + lane16;
      const float bias = (tag < 50) ? out_b[tag] : 0.f;

      v16h bfr[7];
#pragma unroll
      for (int c = 0; c < 7; ++c)
#pragma unroll
        for (int j = 0; j < 8; ++j) {
          const int K = c * 32 + a_kbase(j, hi);
          set_pair(bfr[c], j, *(const uint32_t*)&wT[ntile * 16 + lane16][K]);
        }

      v8f acc;
#pragma unroll
      for (int r = 0; r < 8; ++r) acc[r] = bias;
#pragma unroll
      for (int c = 0; c < 7; ++c) acc = wmma_f16(afr[c], bfr[c], acc);

      if (tag < 50) {
#pragma unroll
        for (int r = 0; r < 8; ++r) {
          const int t = t0 + r + hi * 8;
          out[((size_t)b * 512 + t) * 50 + tag] = acc[r];
        }
      }
    }
  }
}

// ---------------------------------------------------------------------------
extern "C" void kernel_launch(void* const* d_in, const int* in_sizes, int n_in,
                              void* d_out, int out_size, void* d_ws, size_t ws_size,
                              hipStream_t stream) {
  const int*   char_ids = (const int*)d_in[0];
  const int*   text     = (const int*)d_in[1];
  const float* char_emb = (const float*)d_in[2];
  const float* cnn_w    = (const float*)d_in[3];
  const float* cnn_b    = (const float*)d_in[4];
  const float* word_emb = (const float*)d_in[5];
  const float* conv1_w  = (const float*)d_in[6];
  const float* conv1_b  = (const float*)d_in[7];
  const float* conv2_w  = (const float*)d_in[8];
  const float* conv2_b  = (const float*)d_in[9];
  const float* conv3_w  = (const float*)d_in[10];
  const float* conv3_b  = (const float*)d_in[11];
  const float* conv4_w  = (const float*)d_in[12];
  const float* conv4_b  = (const float*)d_in[13];
  const float* out_w    = (const float*)d_in[14];
  const float* out_b    = (const float*)d_in[15];

  // workspace layout (f16): e[65536][160], hA[65536][224], hB[65536][224]
  char* ws = (char*)d_ws;
  const size_t E_BYTES = (size_t)65536 * 160 * sizeof(_Float16);  // 20.97 MB
  const size_t H_BYTES = (size_t)65536 * 224 * sizeof(_Float16);  // 29.36 MB
  _Float16* e  = (_Float16*)ws;
  _Float16* hA = (_Float16*)(ws + E_BYTES);
  _Float16* hB = (_Float16*)(ws + E_BYTES + H_BYTES);

  // 1) char CNN + word embed -> e (65536 words, 1 wave/word)
  char_word_kernel<<<8192, 256, 0, stream>>>(char_ids, text, char_emb,
                                             cnn_w, cnn_b, word_emb, e);

  // 2) conv stack (grid = B * CoPad/16 = 128*14)
  conv_relu_kernel<160><<<1792, 256, 0, stream>>>(e,  conv1_w, conv1_b, hA, 150, 200);
  conv_relu_kernel<224><<<1792, 256, 0, stream>>>(hA, conv2_w, conv2_b, hB, 200, 200);
  conv_relu_kernel<224><<<1792, 256, 0, stream>>>(hB, conv3_w, conv3_b, hA, 200, 200);
  conv_relu_kernel<224><<<1792, 256, 0, stream>>>(hA, conv4_w, conv4_b, hB, 200, 200);

  // 3) projection to tags
  out_proj_kernel<<<128, 256, 0, stream>>>(hB, out_w, out_b, (float*)d_out);
}
